// customConv2_46282567581973
// MI455X (gfx1250) — matrix operand order repaired and checked by reading, not compiled, not run on hardware
//
#include <hip/hip_runtime.h>

// ---------- types for WMMA ----------
typedef __bf16 v16bf __attribute__((ext_vector_type(16)));
typedef float  v8f   __attribute__((ext_vector_type(8)));
typedef unsigned short ush8 __attribute__((ext_vector_type(8)));

#define OC   64
#define IC   64
#define KK   576          // IC * 9, contraction length, reordered as k' = rs*64 + c
#define H    112
#define W_   112
#define TH   8            // output tile rows per workgroup
#define TW   16           // output tile cols per workgroup
#define PTH  10           // padded tile rows
#define PTW  18           // padded tile cols
#define PT   (PTH*PTW)    // 180
#define UELEMS (PT*IC)    // 11520 elements per power buffer

__device__ __forceinline__ unsigned short f2bf(float f) {
    union { float f; unsigned u; } v; v.f = f;
    unsigned r = v.u + 0x7FFFu + ((v.u >> 16) & 1u);   // round-to-nearest-even
    return (unsigned short)(r >> 16);
}

__device__ __forceinline__ float sigm(float x) { return 1.0f / (1.0f + __expf(-x)); }

// ---------------------------------------------------------------------------
// Prelude: W [64,64,3,3] f32  ->  Wbf16 [o][k'] with k' = (r*3+s)*64 + c,
// plus S0[o] = sum_k W[o,k] (for the constant polynomial terms).
// ---------------------------------------------------------------------------
__global__ void prep_weights(const float* __restrict__ w,
                             unsigned short* __restrict__ wbf,
                             float* __restrict__ s0) {
    int o = blockIdx.x;        // 64 blocks
    int c = threadIdx.x;       // 64 threads
    float sum = 0.f;
    #pragma unroll
    for (int rs = 0; rs < 9; ++rs) {
        float v = w[(o * IC + c) * 9 + rs];
        sum += v;
        wbf[o * KK + rs * IC + c] = f2bf(v);
    }
    __shared__ float red[64];
    red[c] = sum;
    __syncthreads();
    if (c == 0) {
        float t = 0.f;
        for (int i = 0; i < 64; ++i) t += red[i];
        s0[o] = t;
    }
}

// ---------------------------------------------------------------------------
// Fused conv-poly kernel. Grid: (W/TW, H/TH, B), 256 threads (8 waves).
// Dynamic LDS: Wl (64x576 bf16) | U1 | U2 | U3 (each PTH*PTW*IC bf16).
// ---------------------------------------------------------------------------
__global__ void __launch_bounds__(256)
conv_poly(const float* __restrict__ img,
          const unsigned short* __restrict__ wbf,
          const float* __restrict__ s0g,
          float* __restrict__ out) {
    extern __shared__ __align__(128) char smem[];
    unsigned short* Wl = (unsigned short*)smem;                       // 73728 B
    unsigned short* U1 = (unsigned short*)(smem + OC * KK * 2);
    unsigned short* U2 = U1 + UELEMS;
    unsigned short* U3 = U2 + UELEMS;

    const int t  = threadIdx.x;
    const int b  = blockIdx.z;
    const int y0 = blockIdx.y * TH;
    const int x0 = blockIdx.x * TW;

    // ---- stage bf16 weights into LDS (coalesced dword copies) ----
    {
        const unsigned int* wsrc = (const unsigned int*)wbf;
        unsigned int*       wdst = (unsigned int*)Wl;
        #pragma unroll 4
        for (int e = t; e < (OC * KK) / 2; e += 256) wdst[e] = wsrc[e];
    }

    // ---- stage padded input tile; compute powers in f32, store bf16 ----
    const float* imb = img + b * IC * H * W_;
    for (int e = t; e < UELEMS; e += 256) {
        int c   = e / PT;
        int pxy = e - c * PT;
        int py  = pxy / PTW;
        int px  = pxy - py * PTW;
        int gy  = y0 + py - 1;
        int gx  = x0 + px - 1;
        float v = 0.f;
        if ((unsigned)gy < (unsigned)H && (unsigned)gx < (unsigned)W_)
            v = imb[(c * H + gy) * W_ + gx];
        float v2 = v * v;
        float v3 = v2 * v;
        int li = pxy * IC + c;        // channels contiguous -> contiguous k' chunks
        U1[li] = f2bf(v);
        U2[li] = f2bf(v2);
        U3[li] = f2bf(v3);
    }
    __syncthreads();

    const int wave = t >> 5;          // 0..7 : output row within tile
    const int lane = t & 31;
    const int nlo  = lane & 15;
    const int hi   = lane >> 4;

    v8f acc[3][4];
    #pragma unroll
    for (int p = 0; p < 3; ++p)
        #pragma unroll
        for (int n = 0; n < 4; ++n)
            acc[p][n] = (v8f){0.f,0.f,0.f,0.f,0.f,0.f,0.f,0.f};

    union AF { v16bf v; ush8 h[2]; };

    // Loop-invariant parts of the fragment addressing.
    // A chunk base for (rs, cb): ((wave+r)*PTW + nlo+s)*IC + cb*32 + hi*8
    const int arow0 = wave * PTW + nlo;       // (r,s) add r*PTW + s
    const int aoff  = hi * 8;
    // B row base: (nt*16 + nlo)*KK, column = rs*64 + cb*32 + hi*8 (+16 for upper half)
    const int brow  = nlo * KK;
    const int boff  = hi * 8;

    // ---- K loop: fully unrolled; per k-chunk issue ALL LDS loads (6 A chunks
    //      + 4 B fragments) before the 12 WMMAs so the scheduler needs a single
    //      dscnt wait per 12 matrix ops instead of one per WMMA-triple ----
    #pragma unroll
    for (int rs = 0; rs < 9; ++rs) {
        const int r = rs / 3, s = rs - r * 3;
        const int abase = (arow0 + r * PTW + s) * IC + aoff;
        const int kb    = rs * IC + boff;
        #pragma unroll
        for (int cb = 0; cb < 2; ++cb) {
            const int ca  = abase + cb * 32;
            const int col = kb + cb * 32;
            AF a1, a2, a3;
            a1.h[0] = *(const ush8*)(U1 + ca);
            a1.h[1] = *(const ush8*)(U1 + ca + 16);
            a2.h[0] = *(const ush8*)(U2 + ca);
            a2.h[1] = *(const ush8*)(U2 + ca + 16);
            a3.h[0] = *(const ush8*)(U3 + ca);
            a3.h[1] = *(const ush8*)(U3 + ca + 16);
            v16bf bfrag[4];
            #pragma unroll
            for (int nt = 0; nt < 4; ++nt) {
                AF bt;
                bt.h[0] = *(const ush8*)(Wl + nt * (16 * KK) + brow + col);
                bt.h[1] = *(const ush8*)(Wl + nt * (16 * KK) + brow + col + 16);
                bfrag[nt] = bt.v;
            }
            #pragma unroll
            for (int nt = 0; nt < 4; ++nt) {
                acc[0][nt] = __builtin_amdgcn_wmma_f32_16x16x32_bf16(
                    false, a1.v, false, bfrag[nt], (short)0, acc[0][nt], false, false);
                acc[1][nt] = __builtin_amdgcn_wmma_f32_16x16x32_bf16(
                    false, a2.v, false, bfrag[nt], (short)0, acc[1][nt], false, false);
                acc[2][nt] = __builtin_amdgcn_wmma_f32_16x16x32_bf16(
                    false, a3.v, false, bfrag[nt], (short)0, acc[2][nt], false, false);
            }
        }
    }

    // ---- epilogue: fold S0 + sigmoid blend, write f32 output ----
    const int y = y0 + wave;
    #pragma unroll
    for (int nt = 0; nt < 4; ++nt) {
        int o = nt * 16 + nlo;
        float S0 = s0g[o];
        float* obase = out + ((b * OC + o) * H + y) * W_ + x0;
        #pragma unroll
        for (int j = 0; j < 8; ++j) {
            // C/D layout: VGPR j holds M=j (lanes 0-15) / M=j+8 (lanes 16-31)
            float u1 = acc[0][nt][j];
            float u2 = acc[1][nt][j];
            float u3 = acc[2][nt][j];
            float f  = (-0.000287f * S0 + 0.266f * u1 - 0.1097f * u2) * (1.0f / 75.0f);
            float g1 = (0.11f  / 75.0f) * 576.0f + 0.001309f  * S0 + 0.00619f  * u1 - 0.009f    * u2 + 0.001383f * u3;
            float g2 = (0.179f / 75.0f) * 576.0f - 0.0025f    * S0 + 0.00303f  * u1 - 0.00484f  * u2 + 0.0175f   * u3;
            float g3 = (0.238f / 75.0f) * 576.0f - 0.000954f  * S0 + 0.00187f  * u1 + 0.001877f * u2 + 0.01502f  * u3;
            float g4 = (0.388f / 75.0f) * 576.0f - 0.00734f   * S0 + 0.001117f * u1 + 0.00752f  * u2 + 0.009f    * u3;
            float g5 = (0.507f / 75.0f) * 576.0f - 0.01017f   * S0 + 0.000426f * u1 + 0.00837f  * u2 + 0.00413f  * u3;
            // s-terms telescope: only 4 sigmoids needed
            float A = sigm(10.f * (0.15f - f));
            float C = sigm(10.f * (0.23f - f));
            float E = sigm(10.f * (0.32f - f));
            float G = sigm(10.f * (0.39f - f));
            float res = A * g1 + (C - A) * g2 + (E - C) * g3 + (G - E) * g4 + (1.f - G) * g5;
            obase[j + hi * 8] = res;
        }
    }
}

extern "C" void kernel_launch(void* const* d_in, const int* in_sizes, int n_in,
                              void* d_out, int out_size, void* d_ws, size_t ws_size,
                              hipStream_t stream) {
    const float* img = (const float*)d_in[0];   // [8,64,112,112]
    const float* wts = (const float*)d_in[1];   // [64,64,3,3]
    unsigned short* wbf = (unsigned short*)d_ws;                   // 64*576 bf16
    float* s0 = (float*)((char*)d_ws + (size_t)OC * KK * sizeof(unsigned short));

    prep_weights<<<64, 64, 0, stream>>>(wts, wbf, s0);

    dim3 grid(W_ / TW, H / TH, 8);               // 7 x 14 x 8 = 784 workgroups
    size_t smem = (size_t)OC * KK * 2 + 3 * (size_t)UELEMS * 2;   // 142848 B
    conv_poly<<<grid, 256, smem, stream>>>(img, wbf, s0, (float*)d_out);
}
